// MultiHeadAttentionBlock_87247965651335
// MI455X (gfx1250) — matrix operand order repaired and checked by reading, not compiled, not run on hardware
//
#include <hip/hip_runtime.h>
#include <hip/hip_bf16.h>

#define HEADS 4
#define Bv 4
#define Cv 256
#define Sv 4096
#define HD 64

typedef __attribute__((ext_vector_type(16))) _Float16 v16h;
typedef __attribute__((ext_vector_type(8)))  _Float16 v8h;
typedef __attribute__((ext_vector_type(8)))  float    v8f;

union V16U { v16h v; v8h h[2]; };

// A-operand fragment (16x32 f16): lane<16 holds row M=lane, K=0..15 contiguous;
// lane>=16 holds row M=lane-16, K=16..31. Caller pre-offsets p by (hi?16:0).
__device__ __forceinline__ v16h load_a_frag(const _Float16* p) {
    V16U u;
    u.h[0] = *(const v8h*)(p);
    u.h[1] = *(const v8h*)(p + 8);
    return u.v;
}
// B-operand fragment (32x16 f16): lane<16 holds col N=lane, K=0..15 contiguous;
// lane>=16 holds col N=lane-16, K=16..31. Caller pre-offsets p by (hi?16:0).
__device__ __forceinline__ v16h load_b_frag(const _Float16* p) {
    V16U u;
    u.h[0] = *(const v8h*)(p);
    u.h[1] = *(const v8h*)(p + 8);
    return u.v;
}

__device__ __forceinline__ v8f wmma_f16(v16h a, v16h b, v8f c) {
    return __builtin_amdgcn_wmma_f32_16x16x32_f16(
        /*neg_a=*/false, a, /*neg_b=*/false, b,
        /*c_mod=*/(short)0, c, /*reuse_a=*/false, /*reuse_b=*/false);
}

// Online-softmax update for one 16-query tile given transposed score tiles
// st0 (keys 0..15) / st1 (keys 16..31). Per lane: query = lane%16, keys in
// components + half-wave partner. Returns the P^T B-operand fragment and
// rescales the Ot accumulators in place.
__device__ __forceinline__ v16h softmax_update(
    v8f st0, v8f st1, float& m_run, float& l_run,
    v8f& o0, v8f& o1, v8f& o2, v8f& o3, bool hi) {
    float mx = -1e30f;
#pragma unroll
    for (int r = 0; r < 8; ++r) { mx = fmaxf(mx, st0[r]); mx = fmaxf(mx, st1[r]); }
    mx = fmaxf(mx, __shfl_xor(mx, 16, 32));
    float m_new = fmaxf(m_run, mx);
    float corr = __expf(m_run - m_new);
    float p0[8], p1[8], sum = 0.0f;
#pragma unroll
    for (int r = 0; r < 8; ++r) {
        p0[r] = __expf(st0[r] - m_new);
        p1[r] = __expf(st1[r] - m_new);
        sum += p0[r] + p1[r];
    }
    sum += __shfl_xor(sum, 16, 32);
    l_run = l_run * corr + sum;
    m_run = m_new;
#pragma unroll
    for (int r = 0; r < 8; ++r) { o0[r] *= corr; o1[r] *= corr; o2[r] *= corr; o3[r] *= corr; }
    v16h pb;
#pragma unroll
    for (int r = 0; r < 8; ++r) {
        float x0 = __shfl_xor(p0[r], 16, 32);
        float x1 = __shfl_xor(p1[r], 16, 32);
        pb[r]     = (_Float16)(hi ? x1 : p0[r]);
        pb[8 + r] = (_Float16)(hi ? p1[r] : x0);
    }
    return pb;
}

// ---------------- prep kernels ----------------

__global__ void k_cvt_f16(const float* __restrict__ in, _Float16* __restrict__ out, int n) {
    int i = blockIdx.x * blockDim.x + threadIdx.x;
    if (i < n) out[i] = (_Float16)in[i];
}

// x: (B, C, S) f32 row-major  ->  Xt: (B*S, C) f16
__global__ void k_prep_x(const float* __restrict__ x, _Float16* __restrict__ xt) {
    int i = blockIdx.x * blockDim.x + threadIdx.x;   // B*C*S = 4,194,304
    int s  = i & (Sv - 1);
    int bc = i >> 12;
    int c  = bc & (Cv - 1);
    int b  = bc >> 8;
    xt[((size_t)(b * Sv + s)) * Cv + c] = (_Float16)x[i];
}

// ---------------- QKV projection GEMM ----------------
// Y(16384 x 768) = Xt(16384 x 256) * Wqkv^T ; scatter to Q/K(b,h,S,hd), V(b,h,hd,S)
__global__ void __launch_bounds__(256) k_qkv(
    const _Float16* __restrict__ xt, const _Float16* __restrict__ wq,
    const float* __restrict__ bq,
    _Float16* __restrict__ Q, _Float16* __restrict__ K, _Float16* __restrict__ V) {
    const int wave = threadIdx.x >> 5;
    const int lane = threadIdx.x & 31;
    const int l16  = lane & 15;
    const bool hi  = lane >= 16;
    int tile = blockIdx.x * 8 + wave;      // 1024 Mtiles * 48 Ntiles
    int nt = tile % 48, mt = tile / 48;
    const int m0 = mt * 16, n0 = nt * 16;

    const _Float16* ap = xt + (size_t)(m0 + l16) * Cv + (hi ? 16 : 0);
    const _Float16* bp = wq + (size_t)(n0 + l16) * Cv + (hi ? 16 : 0);
    v8f acc = {};
#pragma unroll
    for (int c = 0; c < Cv; c += 32) {
        acc = wmma_f16(load_a_frag(ap + c), load_b_frag(bp + c), acc);
    }
    const int n = n0 + l16;                // output column for this lane
    const float bias = bq[n];
#pragma unroll
    for (int r = 0; r < 8; ++r) {
        int m = m0 + r + (hi ? 8 : 0);     // output row (C-layout)
        int b = m >> 12, s = m & (Sv - 1);
        float val = acc[r] + bias;
        if (n < Cv) {                      // Q (scale 1/sqrt(hd) folded in)
            int h = n >> 6, d = n & 63;
            Q[((size_t)((b * HEADS + h) * Sv + s)) * HD + d] = (_Float16)(val * 0.125f);
        } else if (n < 2 * Cv) {           // K
            int nn = n - Cv; int h = nn >> 6, d = nn & 63;
            K[((size_t)((b * HEADS + h) * Sv + s)) * HD + d] = (_Float16)val;
        } else {                           // V transposed: (b,h,hd,S)
            int nn = n - 2 * Cv; int h = nn >> 6, d = nn & 63;
            V[((size_t)((b * HEADS + h) * HD + d)) * Sv + s] = (_Float16)val;
        }
    }
}

// ---------------- flash attention ----------------
// One wave handles 32 queries (two 16-query tiles) of one (b,h), so each K/V
// fragment load feeds 2x the WMMAs (0.5 KB per WMMA of L1 traffic). Scores
// computed transposed (St = K_blk * Q^T) so softmaxed P^T lands directly in
// B-operand layout for Ot += V^T_blk * P^T (half-wave shuffles only).
__global__ void __launch_bounds__(256) k_attn(
    const _Float16* __restrict__ Q, const _Float16* __restrict__ K,
    const _Float16* __restrict__ V, _Float16* __restrict__ out) {
    const int wave = threadIdx.x >> 5;
    const int lane = threadIdx.x & 31;
    const int l16  = lane & 15;
    const bool hi  = lane >= 16;
    int qt = blockIdx.x * 8 + wave;        // 0..2047  (B*H*S/32)
    int bh = qt >> 7;                      // 128 32-query tiles per (b,h)
    int q0 = (qt & 127) << 5;

    const _Float16* Qp = Q + (size_t)bh * Sv * HD;
    const _Float16* Kp = K + (size_t)bh * Sv * HD;
    const _Float16* Vp = V + (size_t)bh * HD * Sv;

    // Q as B-operand (Q^T), 2 chunks over d, for both query tiles
    const _Float16* qbase = Qp + (size_t)(q0 + l16) * HD + (hi ? 16 : 0);
    v16h qA0 = load_b_frag(qbase);
    v16h qA1 = load_b_frag(qbase + 32);
    v16h qB0 = load_b_frag(qbase + 16 * HD);
    v16h qB1 = load_b_frag(qbase + 16 * HD + 32);

    float mA = -1e30f, lA = 0.0f, mB = -1e30f, lB = 0.0f;
    v8f oA0 = {}, oA1 = {}, oA2 = {}, oA3 = {};
    v8f oB0 = {}, oB1 = {}, oB2 = {}, oB3 = {};

    for (int s0 = 0; s0 < Sv; s0 += 32) {
        // K block as A-operand: rows = keys, inner dim = d (2 chunks), 2 key tiles
        const _Float16* kb = Kp + (size_t)(s0 + l16) * HD + (hi ? 16 : 0);
        v16h k00 = load_a_frag(kb);
        v16h k01 = load_a_frag(kb + 32);
        v16h k10 = load_a_frag(kb + 16 * HD);
        v16h k11 = load_a_frag(kb + 16 * HD + 32);

        v8f sA0 = {}, sA1 = {}, sB0 = {}, sB1 = {};
        sA0 = wmma_f16(k00, qA0, sA0); sA0 = wmma_f16(k01, qA1, sA0);
        sA1 = wmma_f16(k10, qA0, sA1); sA1 = wmma_f16(k11, qA1, sA1);
        sB0 = wmma_f16(k00, qB0, sB0); sB0 = wmma_f16(k01, qB1, sB0);
        sB1 = wmma_f16(k10, qB0, sB1); sB1 = wmma_f16(k11, qB1, sB1);

        v16h pbA = softmax_update(sA0, sA1, mA, lA, oA0, oA1, oA2, oA3, hi);
        v16h pbB = softmax_update(sB0, sB1, mB, lB, oB0, oB1, oB2, oB3, hi);

        // Ot(64d x 16q) += V^T_blk(d x 32k) * P^T(32k x 16q); 4 d-tiles, shared V
        const _Float16* vb = Vp + (size_t)l16 * Sv + s0 + (hi ? 16 : 0);
        v16h v0 = load_a_frag(vb);
        v16h v1 = load_a_frag(vb + 16 * (size_t)Sv);
        v16h v2 = load_a_frag(vb + 32 * (size_t)Sv);
        v16h v3 = load_a_frag(vb + 48 * (size_t)Sv);
        oA0 = wmma_f16(v0, pbA, oA0); oA1 = wmma_f16(v1, pbA, oA1);
        oA2 = wmma_f16(v2, pbA, oA2); oA3 = wmma_f16(v3, pbA, oA3);
        oB0 = wmma_f16(v0, pbB, oB0); oB1 = wmma_f16(v1, pbB, oB1);
        oB2 = wmma_f16(v2, pbB, oB2); oB3 = wmma_f16(v3, pbB, oB3);
    }

    float invA = 1.0f / lA;
    float invB = 1.0f / lB;
    int b = bh >> 2, h = bh & 3;
    size_t rowA = (size_t)(b * Sv + q0 + l16) * Cv + h * HD;       // (B*S, C) f16
    size_t rowB = rowA + (size_t)16 * Cv;
#pragma unroll
    for (int r = 0; r < 8; ++r) {
        int dof = r + (hi ? 8 : 0);
        out[rowA + dof]      = (_Float16)(oA0[r] * invA);
        out[rowA + 16 + dof] = (_Float16)(oA1[r] * invA);
        out[rowA + 32 + dof] = (_Float16)(oA2[r] * invA);
        out[rowA + 48 + dof] = (_Float16)(oA3[r] * invA);
        out[rowB + dof]      = (_Float16)(oB0[r] * invB);
        out[rowB + 16 + dof] = (_Float16)(oB1[r] * invB);
        out[rowB + 32 + dof] = (_Float16)(oB2[r] * invB);
        out[rowB + 48 + dof] = (_Float16)(oB3[r] * invB);
    }
}

// ---------------- output projection ----------------
// y(16384 x 256) = AO(16384 x 256) * Wo^T + bo ; store fp32 as (B,C,H,W)
__global__ void __launch_bounds__(256) k_oproj(
    const _Float16* __restrict__ ao, const _Float16* __restrict__ wo,
    const float* __restrict__ bo, float* __restrict__ y) {
    const int wave = threadIdx.x >> 5;
    const int lane = threadIdx.x & 31;
    const int l16  = lane & 15;
    const bool hi  = lane >= 16;
    int tile = blockIdx.x * 8 + wave;      // 1024 Mtiles * 16 Ntiles
    int nt = tile & 15, mt = tile >> 4;
    const int m0 = mt * 16, n0 = nt * 16;

    const _Float16* ap = ao + (size_t)(m0 + l16) * Cv + (hi ? 16 : 0);
    const _Float16* bp = wo + (size_t)(n0 + l16) * Cv + (hi ? 16 : 0);
    v8f acc = {};
#pragma unroll
    for (int c = 0; c < Cv; c += 32) {
        acc = wmma_f16(load_a_frag(ap + c), load_b_frag(bp + c), acc);
    }
    const int n = n0 + l16;
    const float bias = bo[n];
#pragma unroll
    for (int r = 0; r < 8; ++r) {
        int m = m0 + r + (hi ? 8 : 0);
        int b = m >> 12, s = m & (Sv - 1);
        y[((size_t)(b * Cv + n)) * Sv + s] = acc[r] + bias;
    }
}

// ---------------- launch ----------------

extern "C" void kernel_launch(void* const* d_in, const int* in_sizes, int n_in,
                              void* d_out, int out_size, void* d_ws, size_t ws_size,
                              hipStream_t stream) {
    const float* x     = (const float*)d_in[0];
    const float* w_qkv = (const float*)d_in[1];
    const float* b_qkv = (const float*)d_in[2];
    const float* w_o   = (const float*)d_in[3];
    const float* b_o   = (const float*)d_in[4];
    float* y = (float*)d_out;

    char* ws = (char*)d_ws;
    _Float16* Xt = (_Float16*)ws;            ws += (size_t)16384 * 256 * 2;  // 8 MB
    _Float16* Wq = (_Float16*)ws;            ws += (size_t)768 * 256 * 2;
    _Float16* Wo = (_Float16*)ws;            ws += (size_t)256 * 256 * 2;
    _Float16* Qb = (_Float16*)ws;            ws += (size_t)16384 * 64 * 2;   // (b,h,S,hd)
    _Float16* Kb = (_Float16*)ws;            ws += (size_t)16384 * 64 * 2;
    _Float16* Vb = (_Float16*)ws;            ws += (size_t)16384 * 64 * 2;   // (b,h,hd,S)
    _Float16* AO = (_Float16*)ws;            ws += (size_t)16384 * 256 * 2;

    k_cvt_f16<<<768, 256, 0, stream>>>(w_qkv, Wq, 768 * 256);
    k_cvt_f16<<<256, 256, 0, stream>>>(w_o, Wo, 256 * 256);
    k_prep_x<<<16384, 256, 0, stream>>>(x, Xt);
    k_qkv<<<6144, 256, 0, stream>>>(Xt, Wq, b_qkv, Qb, Kb, Vb);
    k_attn<<<256, 256, 0, stream>>>(Qb, Kb, Vb, AO);
    k_oproj<<<2048, 256, 0, stream>>>(AO, Wo, b_o, y);
}
